// RecursiveNN_82162724372982
// MI455X (gfx1250) — matrix operand order repaired and checked by reading, not compiled, not run on hardware
//
#include <hip/hip_runtime.h>

// ---------------------------------------------------------------------------
// Multi-scale neural-CA on MI455X (gfx1250, wave32).
//   state: (2, 12, 256, 256) f32.  Per step:
//     y = depthwise 3x3 conv (ident/sobel_x/sobel_y/lap, wrap pad) -> 48 feat
//     h = relu(W1[i] (96x48) @ y + b1[i]);  out = W2[i] (12x96) @ h
//     state += out
//   10 steps/scale, 3 scales; crop 128x128 + bilinear upsample between scales.
// Per-pixel MLP mapped to v_wmma_f32_16x16x32_bf16:
//   D(16x16) = A(weights 16xK, bf16) x B(features Kx16pixels, bf16) + C(f32)
// Weight A-fragments are pre-packed once per launch into d_ws; bf16 packing
// in the hot loop uses a single v_perm_b32 per pair of floats.
// ---------------------------------------------------------------------------

typedef __attribute__((ext_vector_type(16))) __bf16 v16bf;
typedef __attribute__((ext_vector_type(8)))  float  v8f;

#define Cc      12
#define HH      256
#define WW      256
#define NB      2
#define TILE    16
#define HALO    18                      // TILE + 2
#define PLANE   (HH*WW)
#define STATE_N (NB*Cc*PLANE)
#define NFRAG   15                      // 12 (W1: 6 mtiles x 2 kchunks) + 3 (W2)

__device__ __forceinline__ unsigned short bf16_rne(float f) {
  unsigned int u = __builtin_bit_cast(unsigned int, f);
  u += 0x7FFFu + ((u >> 16) & 1u);
  return (unsigned short)(u >> 16);
}

// pack two f32 into a bf16x2 dword (truncation) with one v_perm_b32:
// result[15:0] = lo[31:16], result[31:16] = hi[31:16]
__device__ __forceinline__ unsigned int bfpack(float hi, float lo) {
  return __builtin_amdgcn_perm(__builtin_bit_cast(unsigned int, hi),
                               __builtin_bit_cast(unsigned int, lo),
                               0x07060302u);
}

// ---------------------------------------------------------------------------
// One-time weight fragment packer (A-matrix 16x32 bf16 layout).
// grid = (NFRAG, 3 scales), block = 32 (one wave builds lane-exact fragments).
// Fragment f < 12: W1, mtile = f/2, kchunk = f%2 (K padded 48 -> 64).
// Fragment f >= 12: W2, kchunk = f-12 (K = 96 = 3 chunks, M padded 12 -> 16).
// Storage: wfrags[((scale*NFRAG + f)*32 + lane)] = 32 bytes (two uint4).
// ---------------------------------------------------------------------------
__global__ __launch_bounds__(32)
void nca_pack_weights(const float* __restrict__ W1,
                      const float* __restrict__ W2,
                      uint4* __restrict__ wfrags)
{
  const int lane = threadIdx.x & 31;
  const int g = lane >> 4, m = lane & 15;
  const int f = blockIdx.x;
  const int i = blockIdx.y;
  const float* W1i = W1 + i * 96 * 48;
  const float* W2i = W2 + i * Cc * 96;

  union { unsigned short s[16]; uint4 q[2]; } u;
  for (int e = 0; e < 16; ++e) {
    // A fragment in this file's contiguous k order (same as B): lanes 0-15 hold K 0..15; lanes 16-31 hold
    // K 16..31 (M = lane%16).
    int klocal = g * 16 + e;
    float v;
    if (f < 12) {
      int mt = f >> 1, kc = f & 1;
      int k = kc * 32 + klocal;
      v = (k < 48) ? W1i[(mt * 16 + m) * 48 + k] : 0.0f;
    } else {
      int kc = f - 12;
      int k = kc * 32 + klocal;
      v = (m < Cc) ? W2i[m * 96 + k] : 0.0f;
    }
    u.s[e] = bf16_rne(v);
  }
  uint4* dst = wfrags + (size_t)((i * NFRAG + f) * 32 + lane) * 2;
  dst[0] = u.q[0];
  dst[1] = u.q[1];
}

__global__ __launch_bounds__(256)
void nca_step(const float* __restrict__ cur, float* __restrict__ nxt,
              const uint4* __restrict__ wfrags,
              const float* __restrict__ b1, int scale_i)
{
  __shared__ float xt[Cc][HALO][HALO];                 // tile + wrap halo
  __shared__ __align__(32) unsigned short h1s[8][16 * 96]; // per-wave, [n][ch]

  const int tid  = threadIdx.x;
  const int lane = tid & 31;
  const int wave = tid >> 5;
  const int g    = lane >> 4;        // lane half (0: lanes 0-15, 1: 16-31)
  const int n    = lane & 15;        // pixel column within strip / N index
  const int tx0  = blockIdx.x * TILE;
  const int ty0  = blockIdx.y * TILE;
  const float* curb = cur + blockIdx.z * (Cc * PLANE);
  float*       nxtb = nxt + blockIdx.z * (Cc * PLANE);

  // ---- stage state tile + 1-pixel wrap halo into LDS -----------------------
  for (int idx = tid; idx < Cc * HALO * HALO; idx += 256) {
    int c  = idx / (HALO * HALO);
    int r  = idx % (HALO * HALO);
    int ly = r / HALO, lx = r % HALO;
    int gy = (ty0 + ly - 1 + HH) & (HH - 1);
    int gx = (tx0 + lx - 1 + WW) & (WW - 1);
    xt[c][ly][lx] = curb[c * PLANE + gy * WW + gx];
  }
  __syncthreads();

  // ---- load pre-packed weight A-fragments (two b128 loads each) ------------
  v16bf aw[NFRAG];
  {
    const uint4* base = wfrags + (size_t)(scale_i * NFRAG * 32 + lane) * 2;
    for (int f = 0; f < NFRAG; ++f) {
      union { v16bf v; uint4 q[2]; } u;
      u.q[0] = base[(size_t)f * 64 + 0];
      u.q[1] = base[(size_t)f * 64 + 1];
      aw[f] = u.v;
    }
  }
  const float* b1i = b1 + scale_i * 96;

  // ---- each wave processes 2 row-strips of 16 pixels -----------------------
  for (int grp = 0; grp < 2; ++grp) {
    const int row = wave * 2 + grp;       // 0..15 within tile
    const int ly  = row + 1;              // LDS coords (halo offset)
    const int lx  = n + 1;

    // perceive -> B fragments (B 32x16 layout: element e at lane l holds
    // row K = kc*32 + g*16 + e, column N = l%16). Feature k -> ch k/4, filt k%4.
    v16bf bfrag[2];
    for (int kc = 0; kc < 2; ++kc) {
      union { v16bf v; unsigned int w[8]; } u;
      for (int cc = 0; cc < 4; ++cc) {
        int c = kc * 8 + g * 4 + cc;      // channel for this group of 4 feats
        if (c >= Cc) {                    // K padding region -> 0
          u.w[cc * 2 + 0] = 0u;
          u.w[cc * 2 + 1] = 0u;
          continue;
        }
        float t00 = xt[c][ly-1][lx-1], t01 = xt[c][ly-1][lx], t02 = xt[c][ly-1][lx+1];
        float t10 = xt[c][ly  ][lx-1], t11 = xt[c][ly  ][lx], t12 = xt[c][ly  ][lx+1];
        float t20 = xt[c][ly+1][lx-1], t21 = xt[c][ly+1][lx], t22 = xt[c][ly+1][lx+1];
        float ident = t11;
        float sx  = (t02 - t00) + 2.0f * (t12 - t10) + (t22 - t20);
        float sy  = (t20 - t00) + 2.0f * (t21 - t01) + (t22 - t02);
        float lap = t00 + 2.0f*t01 + t02 + 2.0f*t10 - 12.0f*t11 + 2.0f*t12
                  + t20 + 2.0f*t21 + t22;
        u.w[cc * 2 + 0] = bfpack(sx, ident);   // elements cc*4+0, cc*4+1
        u.w[cc * 2 + 1] = bfpack(lap, sy);     // elements cc*4+2, cc*4+3
      }
      bfrag[kc] = u.v;
    }

    // ---- layer 1: h1(96 x 16px) = W1 x y, bias + relu, stash in LDS --------
    // h1s layout is pixel-major: h1s[wave][n*96 + ch]  -> packed b32 stores,
    // contiguous b128 reads for the layer-2 B fragment.
    for (int mt = 0; mt < 6; ++mt) {
      v8f acc = {};
      acc = __builtin_amdgcn_wmma_f32_16x16x32_bf16(false, aw[mt * 2 + 0], false,
                                                    bfrag[0], (short)0, acc,
                                                    false, false);
      acc = __builtin_amdgcn_wmma_f32_16x16x32_bf16(false, aw[mt * 2 + 1], false,
                                                    bfrag[1], (short)0, acc,
                                                    false, false);
      // D layout: VGPR r -> channel mt*16 + g*8 + r, pixel column n
      const int ch0 = mt * 16 + g * 8;
      for (int j = 0; j < 4; ++j) {
        float h0 = fmaxf(acc[2 * j + 0] + b1i[ch0 + 2 * j + 0], 0.0f);
        float h1 = fmaxf(acc[2 * j + 1] + b1i[ch0 + 2 * j + 1], 0.0f);
        *(unsigned int*)&h1s[wave][n * 96 + ch0 + 2 * j] = bfpack(h1, h0);
      }
    }
    // wave-local LDS RAW: drain DS counter before cross-lane re-layout read
    asm volatile("s_wait_dscnt 0" ::: "memory");

    // ---- layer 2: out(12 x 16px) = W2 x h1 ---------------------------------
    v8f acc2 = {};
    for (int kc = 0; kc < 3; ++kc) {
      // B element e at lane l = h1[K = kc*32 + g*16 + e][pixel n]:
      // contiguous 32 bytes in pixel-major h1s -> single v16bf LDS load.
      const v16bf bv =
          *(const v16bf*)&h1s[wave][n * 96 + kc * 32 + g * 16];
      acc2 = __builtin_amdgcn_wmma_f32_16x16x32_bf16(false, aw[12 + kc], false,
                                                     bv, (short)0, acc2,
                                                     false, false);
    }

    // residual write: lane covers channels g*8 + r (valid < 12) at (row, n)
    const int gy = ty0 + row, gx = tx0 + n;
    for (int r = 0; r < 8; ++r) {
      int ch = g * 8 + r;
      if (ch < Cc) {
        int off = ch * PLANE + gy * WW + gx;
        nxtb[off] = curb[off] + acc2[r];
      }
    }
  }
}

// Bilinear upsample of the top-left 128x128 crop back to 256x256
// (half-pixel centers, matching jax.image.resize 'bilinear').
__global__ __launch_bounds__(256)
void nca_resize(const float* __restrict__ src, float* __restrict__ dst)
{
  int idx = blockIdx.x * 256 + threadIdx.x;
  if (idx >= STATE_N) return;
  int x = idx & 255;
  int y = (idx >> 8) & 255;
  int plane = idx >> 16;                      // b*C + c
  const float* s = src + plane * PLANE;

  float fx = x * 0.5f - 0.25f;
  float fy = y * 0.5f - 0.25f;
  fx = fx < 0.0f ? 0.0f : (fx > 127.0f ? 127.0f : fx);
  fy = fy < 0.0f ? 0.0f : (fy > 127.0f ? 127.0f : fy);
  int x0 = (int)fx, y0 = (int)fy;
  int x1 = x0 + 1 > 127 ? 127 : x0 + 1;
  int y1 = y0 + 1 > 127 ? 127 : y0 + 1;
  float wx = fx - (float)x0, wy = fy - (float)y0;

  float v00 = s[y0 * WW + x0], v01 = s[y0 * WW + x1];
  float v10 = s[y1 * WW + x0], v11 = s[y1 * WW + x1];
  float top = v00 + wx * (v01 - v00);
  float bot = v10 + wx * (v11 - v10);
  dst[idx] = top + wy * (bot - top);
}

extern "C" void kernel_launch(void* const* d_in, const int* in_sizes, int n_in,
                              void* d_out, int out_size, void* d_ws, size_t ws_size,
                              hipStream_t stream)
{
  const float* x  = (const float*)d_in[0];   // (2,12,256,256)
  const float* W1 = (const float*)d_in[1];   // (3,96,48)
  const float* b1 = (const float*)d_in[2];   // (3,96)
  const float* W2 = (const float*)d_in[3];   // (3,12,96)
  float* out = (float*)d_out;                // (3,2,12,256,256)

  float* s0 = (float*)d_ws;                  // ping
  float* s1 = s0 + STATE_N;                  // pong
  uint4* wfrags = (uint4*)(s1 + STATE_N);    // 3*15*32*32 B = 45 KB
  const size_t bytes = (size_t)STATE_N * sizeof(float);

  nca_pack_weights<<<dim3(NFRAG, 3), 32, 0, stream>>>(W1, W2, wfrags);
  hipMemcpyAsync(s0, x, bytes, hipMemcpyDeviceToDevice, stream);

  float* curp = s0;
  float* nxtp = s1;
  dim3 grid(WW / TILE, HH / TILE, NB), block(256);

  for (int i = 0; i < 3; ++i) {
    for (int t = 0; t < 10; ++t) {
      nca_step<<<grid, block, 0, stream>>>(curp, nxtp, wfrags, b1, i);
      float* tmp = curp; curp = nxtp; nxtp = tmp;
    }
    hipMemcpyAsync(out + (size_t)i * STATE_N, curp, bytes,
                   hipMemcpyDeviceToDevice, stream);
    if (i < 2) {
      nca_resize<<<(STATE_N + 255) / 256, 256, 0, stream>>>(curp, nxtp);
      float* tmp = curp; curp = nxtp; nxtp = tmp;
    }
  }
}
